// HungarianMSELoss_16862041604809
// MI455X (gfx1250) — compile-verified, matrix operand order checked by reading, not run on hardware
//
#include <hip/hip_runtime.h>
#include <math.h>

typedef __attribute__((ext_vector_type(16))) _Float16 v16h;
typedef __attribute__((ext_vector_type(8)))  float    v8f;

#define NPTS 256
#define NP1  257
#define BIGF 1e30f

// All solver state lives in LDS: 256KB cost matrix + coords + JV arrays.
// Total ~275KB < 320KB per-WGP LDS on CDNA5 (one workgroup per WGP).
struct Smem {
    float cost[NPTS * NPTS];            // 262144 B
    float px[NPTS], py[NPTS], tx[NPTS], ty[NPTS];
    float pn2[NPTS], tn2[NPTS];         // squared norms
    float u[NP1], v[NP1], minv[NP1];
    int   p[NP1], way[NP1], used[NP1];
    float redV[8]; int redI[8];
    float deltaS; int j0S, j1S;
};

__global__ __launch_bounds__(256)
void hungarian_kernel(const float* __restrict__ pred,
                      const float* __restrict__ target,
                      float* __restrict__ partial)
{
    extern __shared__ char raw[];
    Smem& S = *reinterpret_cast<Smem*>(raw);

    const int tid  = threadIdx.x;
    const int lane = tid & 31;
    const int wid  = tid >> 5;
    const int b    = blockIdx.x;

    const float* pr = pred   + (size_t)b * (NPTS * 2);
    const float* tg = target + (size_t)b * (NPTS * 2);
    __builtin_prefetch(pr, 0, 0);   // global_prefetch_b8
    __builtin_prefetch(tg, 0, 0);

    // ---- stage coordinates + squared norms into LDS ----
    {
        float x = pr[2 * tid], y = pr[2 * tid + 1];
        S.px[tid] = x; S.py[tid] = y; S.pn2[tid] = x * x + y * y;
        float xt = tg[2 * tid], yt = tg[2 * tid + 1];
        S.tx[tid] = xt; S.ty[tid] = yt; S.tn2[tid] = xt * xt + yt * yt;
    }
    __syncthreads();

    // ---- cost matrix via WMMA: dist^2 = |p|^2 + |t|^2 - 2 p.t ----
    // P.T^T computed with v_wmma_f32_16x16x32_f16 using split-f16:
    // K slots 0..5 of A = [phx,phy,plx,ply,phx,phy]
    // K slots 0..5 of B = [thx,thy,thx,thy,tlx,tly]
    // => dot = ph.th + pl.th + ph.tl  (~f32 accuracy)
    // 16x16 grid of 16x16 tiles, 8 waves -> 32 tiles each.
    #pragma unroll 1
    for (int t = wid; t < 256; t += 8) {
        const int m0 = (t >> 4) << 4;
        const int n0 = (t & 15) << 4;
        v16h a  = {};
        v16h bb = {};
        if (lane < 16) {
            float axf = S.px[m0 + lane], ayf = S.py[m0 + lane];
            _Float16 axh = (_Float16)axf, ayh = (_Float16)ayf;
            _Float16 axl = (_Float16)(axf - (float)axh);
            _Float16 ayl = (_Float16)(ayf - (float)ayh);
            a[0] = axh; a[1] = ayh; a[2] = axl; a[3] = ayl; a[4] = axh; a[5] = ayh;

            float bxf = S.tx[n0 + lane], byf = S.ty[n0 + lane];
            _Float16 bxh = (_Float16)bxf, byh = (_Float16)byf;
            _Float16 bxl = (_Float16)(bxf - (float)bxh);
            _Float16 byl = (_Float16)(byf - (float)byh);
            bb[0] = bxh; bb[1] = byh; bb[2] = bxh; bb[3] = byh; bb[4] = bxl; bb[5] = byl;
        }
        v8f c = {};
        c = __builtin_amdgcn_wmma_f32_16x16x32_f16(
                false, a, false, bb, (short)0, c, false, false);

        const int mBase = m0 + ((lane >> 4) << 3);   // lanes 16..31 -> M+8
        const int nn    = n0 + (lane & 15);
        #pragma unroll
        for (int r = 0; r < 8; ++r) {
            const int mm = mBase + r;
            float d2 = S.pn2[mm] + S.tn2[nn] - 2.0f * c[r];
            S.cost[mm * NPTS + nn] = sqrtf(fmaxf(d2, 0.0f));
        }
    }

    // ---- JV / Hungarian init ----
    if (tid == 0) { S.u[0] = 0.0f; S.v[0] = 0.0f; S.p[0] = 0; S.way[0] = 0; }
    S.u[tid + 1] = 0.0f; S.v[tid + 1] = 0.0f; S.p[tid + 1] = 0; S.way[tid + 1] = 0;
    __syncthreads();

    // ---- main loop: insert each row via shortest augmenting path ----
    for (int i = 1; i <= NPTS; ++i) {
        if (tid == 0) { S.p[0] = i; S.j0S = 0; S.used[0] = 0; S.minv[0] = BIGF; }
        S.minv[tid + 1] = BIGF;
        S.used[tid + 1] = 0;
        __syncthreads();

        int j1 = 0;
        for (;;) {
            const int j0 = S.j0S;
            if (tid == 0) S.used[j0] = 1;
            __syncthreads();

            const int   i0   = S.p[j0];
            const float cur  = S.cost[(i0 - 1) * NPTS + tid] - S.u[i0] - S.v[tid + 1];
            const int   freeJ = (S.used[tid + 1] == 0);
            float mv = S.minv[tid + 1];
            if (freeJ && cur < mv) {
                mv = cur;
                S.minv[tid + 1] = mv;
                S.way[tid + 1]  = j0;
            }
            float cand = freeJ ? mv : BIGF;
            int   idx  = tid + 1;

            // wave32 argmin (first-index tie-break like np.argmin)
            #pragma unroll
            for (int off = 16; off > 0; off >>= 1) {
                float v2 = __shfl_down(cand, off, 32);
                int   i2 = __shfl_down(idx,  off, 32);
                if (v2 < cand || (v2 == cand && i2 < idx)) { cand = v2; idx = i2; }
            }
            if (lane == 0) { S.redV[wid] = cand; S.redI[wid] = idx; }
            __syncthreads();
            if (tid == 0) {
                float bv = S.redV[0]; int bi = S.redI[0];
                #pragma unroll
                for (int w = 1; w < 8; ++w) {
                    float v2 = S.redV[w]; int i2 = S.redI[w];
                    if (v2 < bv || (v2 == bv && i2 < bi)) { bv = v2; bi = i2; }
                }
                S.deltaS = bv; S.j1S = bi;
            }
            __syncthreads();

            const float delta = S.deltaS;
            j1 = S.j1S;

            // dual updates (disjoint writes: matched rows are distinct)
            {
                const int j = tid + 1;
                if (S.used[j]) { S.u[S.p[j]] += delta; S.v[j] -= delta; }
                else           { S.minv[j]  -= delta; }
                if (tid == 0 && S.used[0]) { S.u[S.p[0]] += delta; S.v[0] -= delta; }
            }
            __syncthreads();
            if (tid == 0) S.j0S = j1;
            __syncthreads();
            if (S.p[j1] == 0) break;
        }

        // augment along the path (serial, short)
        if (tid == 0) {
            int j0 = j1;
            while (j0) { const int jp = S.way[j0]; S.p[j0] = S.p[jp]; j0 = jp; }
        }
        __syncthreads();
    }

    // ---- per-batch SSE: column j=tid matched to row p[tid+1]-1 ----
    {
        const int r  = S.p[tid + 1] - 1;
        const float dx = S.px[r] - S.tx[tid];
        const float dy = S.py[r] - S.ty[tid];
        float s = dx * dx + dy * dy;
        #pragma unroll
        for (int off = 16; off > 0; off >>= 1) s += __shfl_down(s, off, 32);
        if (lane == 0) S.redV[wid] = s;
        __syncthreads();
        if (tid == 0) {
            float tot = 0.0f;
            #pragma unroll
            for (int w = 0; w < 8; ++w) tot += S.redV[w];
            partial[b] = tot;
        }
    }
}

__global__ void finalize_kernel(const float* __restrict__ partial,
                                float* __restrict__ out, int B)
{
    const int tid = threadIdx.x;
    float s = (tid < B) ? partial[tid] : 0.0f;
    #pragma unroll
    for (int off = 16; off > 0; off >>= 1) s += __shfl_down(s, off, 32);
    if (tid == 0) out[0] = s / (float)(B * NPTS * 2);
}

extern "C" void kernel_launch(void* const* d_in, const int* in_sizes, int n_in,
                              void* d_out, int out_size, void* d_ws, size_t ws_size,
                              hipStream_t stream)
{
    const float* pred   = (const float*)d_in[0];
    const float* target = (const float*)d_in[1];
    float* out     = (float*)d_out;
    float* partial = (float*)d_ws;

    const int B = in_sizes[0] / (NPTS * 2);   // 32 for the reference shapes
    const size_t smem = sizeof(Smem);         // ~275 KB dynamic LDS (CDNA5: 320KB/WGP)

    (void)hipFuncSetAttribute((const void*)hungarian_kernel,
                              hipFuncAttributeMaxDynamicSharedMemorySize,
                              (int)smem);

    hungarian_kernel<<<B, 256, smem, stream>>>(pred, target, partial);
    finalize_kernel<<<1, 32, 0, stream>>>(partial, out, B);
}
